// DualContextAttention_7464653160637
// MI455X (gfx1250) — matrix operand order repaired and checked by reading, not compiled, not run on hardware
//
#include <hip/hip_runtime.h>

#define DIMN 1024
#define HEADSN 16
#define HD 64
#define BB 2
#define SS 2048
#define LL 2048

typedef unsigned short u16;
typedef __bf16 bf16;
typedef bf16 v16bf __attribute__((ext_vector_type(16)));
typedef u16  v16u  __attribute__((ext_vector_type(16)));
typedef u16  v8u   __attribute__((ext_vector_type(8)));
typedef u16  v4u   __attribute__((ext_vector_type(4)));
typedef float v8f  __attribute__((ext_vector_type(8)));

// ---------- helpers ----------
static __device__ __forceinline__ u16 f2bf(float f) {
  unsigned x = __builtin_bit_cast(unsigned, f);
  unsigned r = x + 0x7FFFu + ((x >> 16) & 1u);   // round-to-nearest-even
  return (u16)(r >> 16);
}

static __device__ __forceinline__ v8u cvt8(v8f f) {
  v8u r;
#pragma unroll
  for (int i = 0; i < 8; ++i) r[i] = f2bf(f[i]);
  return r;
}

static __device__ __forceinline__ v16bf mk16(v8u lo, v8u hi) {
  v16u t = __builtin_shufflevector(lo, hi, 0,1,2,3,4,5,6,7,8,9,10,11,12,13,14,15);
  return __builtin_bit_cast(v16bf, t);
}

static __device__ __forceinline__ v16bf asbf(v16u u) {
  return __builtin_bit_cast(v16bf, u);
}

static __device__ __forceinline__ v8f wmma_bf16(v16bf a, v16bf b, v8f c) {
  return __builtin_amdgcn_wmma_f32_16x16x32_bf16(false, a, false, b, (short)0, c, false, false);
}

static __device__ __forceinline__ float wred_max(float v) {
#pragma unroll
  for (int o = 16; o > 0; o >>= 1) v = fmaxf(v, __shfl_xor(v, o, 32));
  return v;
}
static __device__ __forceinline__ float wred_sum(float v) {
#pragma unroll
  for (int o = 16; o > 0; o >>= 1) v += __shfl_xor(v, o, 32);
  return v;
}

// ---------- kernel 0: fp32 -> bf16 bulk convert ----------
__global__ void __launch_bounds__(256)
cvt_f32_bf16_kernel(const float* __restrict__ src, u16* __restrict__ dst, int n4) {
  int i = blockIdx.x * 256 + threadIdx.x;
  if (i < n4) {
    float4 f = ((const float4*)src)[i];
    v4u r;
    r[0] = f2bf(f.x); r[1] = f2bf(f.y); r[2] = f2bf(f.z); r[3] = f2bf(f.w);
    ((v4u*)dst)[i] = r;
  }
}

// ---------- kernel 1: q = (query @ Wq^T + bq) * hd^-0.5, stored bf16 ----------
// M=B*S=4096, N=DIM, K=DIM. Workgroup = 8 waves -> 64x128 C tile.
__global__ void __launch_bounds__(256)
qproj_kernel(const float* __restrict__ query, const u16* __restrict__ wqb,
             const float* __restrict__ bq, u16* __restrict__ qb) {
  const int lane = threadIdx.x & 31, wave = threadIdx.x >> 5;
  const int hlf = lane >> 4, nl = lane & 15;
  const int mrow = (blockIdx.x >> 3) * 64 + (wave & 3) * 16;
  const int ncol = (blockIdx.x & 7) * 128 + (wave >> 2) * 64;
  const float* arow = query + (size_t)(mrow + nl) * DIMN;

  v8f acc[4] = {};
  for (int kk = 0; kk < DIMN; kk += 32) {
    v8f f0 = *(const v8f*)(arow + kk + hlf * 16);
    v8f f1 = *(const v8f*)(arow + kk + hlf * 16 + 8);
    v16bf A = mk16(cvt8(f0), cvt8(f1));
#pragma unroll
    for (int t = 0; t < 4; ++t) {
      const u16* brow = wqb + (size_t)(ncol + t * 16 + nl) * DIMN + kk + hlf * 16;
      acc[t] = wmma_bf16(A, asbf(*(const v16u*)brow), acc[t]);
    }
  }
#pragma unroll
  for (int t = 0; t < 4; ++t) {
    const int n = ncol + t * 16 + nl;
    const float bias = bq[n];
#pragma unroll
    for (int r = 0; r < 8; ++r) {
      const int m = mrow + r + hlf * 8;
      qb[(size_t)m * DIMN + n] = f2bf((acc[t][r] + bias) * 0.125f);
    }
  }
}

// ---------- kernel 2: dual-context attention ----------
// One workgroup per (b, 16-row s tile); loops over contexts and heads.
// LDS: scores 16x2048 f32 | head-mean accum 16x2048 f32 | 16x64 reduction tile.
__global__ void __launch_bounds__(256)
attn_kernel(const u16* __restrict__ qb,
            const u16* __restrict__ k0b, const u16* __restrict__ v0b,
            const u16* __restrict__ k1b, const u16* __restrict__ v1b,
            const int* __restrict__ mask, const float* __restrict__ gatep,
            float* __restrict__ comb, float* __restrict__ a0, float* __restrict__ a1) {
  extern __shared__ float smem[];
  float* sc   = smem;                 // 16*2048
  float* asum = smem + 16 * LL;       // 16*2048
  float* red  = smem + 32 * LL;       // 16*64
  const int tid = threadIdx.x, lane = tid & 31, wave = tid >> 5;
  const int hlf = lane >> 4, nl = lane & 15;
  const int b  = blockIdx.x >> 7;        // S/16 = 128 tiles
  const int s0 = (blockIdx.x & 127) * 16;
  const float g = 1.0f / (1.0f + __expf(-gatep[0]));

  for (int c = 0; c < 2; ++c) {
    const u16* kp = c ? k1b : k0b;
    const u16* vp = c ? v1b : v0b;
    const float cw = c ? (1.0f - g) : g;
    float* aout = c ? a1 : a0;

    for (int h = 0; h < HEADSN; ++h) {
      // Q fragments for this head (scale already folded in)
      const u16* qrow = qb + ((size_t)(b * SS + s0 + nl)) * DIMN + h * HD;
      v16bf Alo = mk16(*(const v8u*)(qrow + hlf * 16),      *(const v8u*)(qrow + hlf * 16 + 8));
      v16bf Ahi = mk16(*(const v8u*)(qrow + 32 + hlf * 16), *(const v8u*)(qrow + 40 + hlf * 16));

      // ---- scores = q k^T : each wave covers 16 tiles of 16 keys ----
      for (int i = 0; i < 16; ++i) {
        const int kb0 = (wave * 16 + i) * 16;
        const u16* krow = kp + ((size_t)(b * LL + kb0 + nl)) * DIMN + h * HD + hlf * 16;
        v8f cacc = {};
        cacc = wmma_bf16(Alo, asbf(*(const v16u*)krow),        cacc);
        cacc = wmma_bf16(Ahi, asbf(*(const v16u*)(krow + 32)), cacc);
#pragma unroll
        for (int r = 0; r < 8; ++r)
          sc[(r + hlf * 8) * LL + kb0 + nl] = cacc[r];
      }
      __syncthreads();

      // ---- exact softmax over full rows; head-mean accumulation ----
#pragma unroll
      for (int rr = 0; rr < 2; ++rr) {
        const int m = wave * 2 + rr;
        float* row = sc + m * LL;
        const int* mrow = mask + (size_t)(s0 + m) * LL;
        float mx = -3.0e38f;
        for (int j = lane; j < LL; j += 32) {
          float v = row[j];
          if (c == 0 && mrow[j] == 0) { v = -3.0e38f; row[j] = v; }
          mx = fmaxf(mx, v);
        }
        mx = wred_max(mx);
        float sum = 0.f;
        for (int j = lane; j < LL; j += 32) {
          float e = __expf(row[j] - mx);
          row[j] = e; sum += e;
        }
        sum = wred_sum(sum);
        const float inv = 1.0f / sum;
        float* arow2 = asum + m * LL;
        for (int j = lane; j < LL; j += 32) {
          float p = row[j] * inv;
          row[j] = p;
          float adds = p * (1.0f / HEADSN);
          arow2[j] = (h == 0) ? adds : (arow2[j] + adds);
        }
      }
      for (int i = tid; i < 16 * 64; i += 256) red[i] = 0.f;
      __syncthreads();

      // ---- P @ V : K split across waves (256 keys each), LDS reduction ----
      v8f ov[4] = {};
      for (int kk = 0; kk < 256; kk += 32) {
        const int kbase = wave * 256 + kk;
        v16u pu;
        const float* prow = sc + nl * LL + kbase + hlf * 16;
#pragma unroll
        for (int j = 0; j < 8; ++j) pu[j]     = f2bf(prow[j]);
#pragma unroll
        for (int j = 0; j < 8; ++j) pu[j + 8] = f2bf(prow[8 + j]);
        v16bf PA = asbf(pu);
#pragma unroll
        for (int t = 0; t < 4; ++t) {
          v16u vu;
          const u16* vbase = vp + ((size_t)(b * LL + kbase + hlf * 16)) * DIMN + h * HD + t * 16 + nl;
#pragma unroll
          for (int e = 0; e < 16; ++e) vu[e] = vbase[(size_t)e * DIMN];
          ov[t] = wmma_bf16(PA, asbf(vu), ov[t]);
        }
      }
#pragma unroll
      for (int t = 0; t < 4; ++t)
#pragma unroll
        for (int r = 0; r < 8; ++r)
          atomicAdd(&red[(r + hlf * 8) * 64 + t * 16 + nl], ov[t][r] * cw);
      __syncthreads();

      // ---- gated accumulate into global comb buffer (c==0 writes, c==1 adds) ----
      for (int i = tid; i < 16 * 64; i += 256) {
        const int m = i >> 6, d = i & 63;
        float* dst = comb + ((size_t)(b * SS + s0 + m)) * DIMN + h * HD + d;
        if (c == 0) *dst = red[i]; else *dst += red[i];
      }
      __syncthreads();
    } // heads

    // ---- write head-mean attention map (each element written exactly once) ----
    for (int i = tid; i < 16 * LL; i += 256) {
      const int m = i >> 11, col = i & (LL - 1);
      aout[((size_t)(b * SS + s0 + m)) * LL + col] = asum[i];
    }
    __syncthreads();
  } // contexts
}

// ---------- kernel 3: out = comb @ Wo^T + bo (fp32 out) ----------
__global__ void __launch_bounds__(256)
oproj_kernel(const float* __restrict__ comb, const u16* __restrict__ wob,
             const float* __restrict__ bo, float* __restrict__ out) {
  const int lane = threadIdx.x & 31, wave = threadIdx.x >> 5;
  const int hlf = lane >> 4, nl = lane & 15;
  const int mrow = (blockIdx.x >> 3) * 64 + (wave & 3) * 16;
  const int ncol = (blockIdx.x & 7) * 128 + (wave >> 2) * 64;
  const float* arow = comb + (size_t)(mrow + nl) * DIMN;

  v8f acc[4] = {};
  for (int kk = 0; kk < DIMN; kk += 32) {
    v8f f0 = *(const v8f*)(arow + kk + hlf * 16);
    v8f f1 = *(const v8f*)(arow + kk + hlf * 16 + 8);
    v16bf A = mk16(cvt8(f0), cvt8(f1));
#pragma unroll
    for (int t = 0; t < 4; ++t) {
      const u16* brow = wob + (size_t)(ncol + t * 16 + nl) * DIMN + kk + hlf * 16;
      acc[t] = wmma_bf16(A, asbf(*(const v16u*)brow), acc[t]);
    }
  }
#pragma unroll
  for (int t = 0; t < 4; ++t) {
    const int n = ncol + t * 16 + nl;
    const float bias = bo[n];
#pragma unroll
    for (int r = 0; r < 8; ++r) {
      const int m = mrow + r + hlf * 8;
      out[(size_t)m * DIMN + n] = acc[t][r] + bias;
    }
  }
}

extern "C" void kernel_launch(void* const* d_in, const int* in_sizes, int n_in,
                              void* d_out, int out_size, void* d_ws, size_t ws_size,
                              hipStream_t stream) {
  (void)in_sizes; (void)n_in; (void)out_size; (void)ws_size;
  const float* query = (const float*)d_in[0];
  const float* k0    = (const float*)d_in[1];
  const float* v0    = (const float*)d_in[2];
  const float* k1    = (const float*)d_in[3];
  const float* v1    = (const float*)d_in[4];
  const int*   msk   = (const int*)d_in[5];
  const float* Wq    = (const float*)d_in[6];
  const float* bq    = (const float*)d_in[7];
  const float* Wo    = (const float*)d_in[8];
  const float* bo    = (const float*)d_in[9];
  const float* gate  = (const float*)d_in[10];

  char* ws = (char*)d_ws;
  u16*   wqb  = (u16*)(ws);                          // 2 MB
  u16*   wob  = (u16*)(ws + (size_t)( 2u << 20));    // 2 MB
  u16*   qb   = (u16*)(ws + (size_t)( 4u << 20));    // 8 MB
  u16*   k0b  = (u16*)(ws + (size_t)(12u << 20));    // 8 MB
  u16*   v0b  = (u16*)(ws + (size_t)(20u << 20));    // 8 MB
  u16*   k1b  = (u16*)(ws + (size_t)(28u << 20));    // 8 MB
  u16*   v1b  = (u16*)(ws + (size_t)(36u << 20));    // 8 MB
  float* comb = (float*)(ws + (size_t)(44u << 20));  // 16 MB

  float* out = (float*)d_out;
  float* a0  = out + (size_t)BB * SS * DIMN;
  float* a1  = a0 + (size_t)BB * SS * LL;

  const int nW  = DIMN * DIMN;        // 1M
  const int nKV = BB * LL * DIMN;     // 4M
  cvt_f32_bf16_kernel<<<(nW  / 4 + 255) / 256, 256, 0, stream>>>(Wq, wqb, nW  / 4);
  cvt_f32_bf16_kernel<<<(nW  / 4 + 255) / 256, 256, 0, stream>>>(Wo, wob, nW  / 4);
  cvt_f32_bf16_kernel<<<(nKV / 4 + 255) / 256, 256, 0, stream>>>(k0, k0b, nKV / 4);
  cvt_f32_bf16_kernel<<<(nKV / 4 + 255) / 256, 256, 0, stream>>>(v0, v0b, nKV / 4);
  cvt_f32_bf16_kernel<<<(nKV / 4 + 255) / 256, 256, 0, stream>>>(k1, k1b, nKV / 4);
  cvt_f32_bf16_kernel<<<(nKV / 4 + 255) / 256, 256, 0, stream>>>(v1, v1b, nKV / 4);

  qproj_kernel<<<512, 256, 0, stream>>>(query, wqb, bq, qb);

  const size_t ldsBytes = (size_t)(16 * LL + 16 * LL + 16 * 64) * sizeof(float); // 260 KB
  attn_kernel<<<BB * (SS / 16), 256, ldsBytes, stream>>>(qb, k0b, v0b, k1b, v1b,
                                                         msk, gate, comb, a0, a1);

  oproj_kernel<<<512, 256, 0, stream>>>(comb, wob, bo, out);
}
